// AttentionLayer_42777874268335
// MI455X (gfx1250) — matrix operand order repaired and checked by reading, not compiled, not run on hardware
//
#include <hip/hip_runtime.h>
#include <hip/hip_bf16.h>

typedef __attribute__((ext_vector_type(16))) _Float16 v16h;
typedef __attribute__((ext_vector_type(8)))  _Float16 v8h;
typedef __attribute__((ext_vector_type(8)))  float    v8f;

#define Bn 4096
#define Tn 200
#define Dn 64
#define H1n 80
#define H2n 40
#define TTn 208              // T padded to 13 tiles of 16
#define NT1 5                // 80/16 column tiles, layer 1
#define KC1 4                // 128/32 K chunks, layer 1
#define NT2 3                // 48/16 column tiles, layer 2
#define KC2 3                // 96/32 K chunks, layer 2
#define H1S 104              // padded h1 row stride (halves), bank-friendly

__device__ __forceinline__ float sigmoidf_(float x) {
    return 1.0f / (1.0f + __expf(-x));
}

__global__ __launch_bounds__(256) void din_attn_kernel(
    const float* __restrict__ q,  const float* __restrict__ k,
    const float* __restrict__ v,  const int*   __restrict__ mask,
    const float* __restrict__ W1, const float* __restrict__ b1,
    const float* __restrict__ W2, const float* __restrict__ b2,
    const float* __restrict__ Wf, const float* __restrict__ bf,
    float* __restrict__ out)
{
    __shared__ __attribute__((aligned(32))) _Float16 sh_w1f[NT1 * KC1 * 32 * 16]; // 20 KB
    __shared__ __attribute__((aligned(32))) _Float16 sh_w2f[NT2 * KC2 * 32 * 16]; //  9 KB
    __shared__ __attribute__((aligned(32))) _Float16 sh_h1[8 * 16 * H1S];         // 26 KB
    __shared__ float sh_c[H1n];
    __shared__ float sh_q[Dn];
    __shared__ float sh_logits[TTn];
    __shared__ float sh_sw[TTn];
    __shared__ float sh_red[8];
    __shared__ float sh_out[4 * Dn];

    const int bid     = blockIdx.x;
    const int tid     = threadIdx.x;
    const int lane    = tid & 31;
    const int wid     = tid >> 5;
    const int lanelow = lane & 15;
    const int hi      = lane >> 4;

    // ---------------- prologue: weights -> LDS fragment layout ----------------
    if (tid < Dn) sh_q[tid] = q[(size_t)bid * Dn + tid];

    // layer-1 combined weights: Wc[K][N], K<64 -> W1k-W1m, K>=64 -> W1p
    for (int f = tid; f < NT1 * KC1 * 32 * 16; f += 256) {
        int h  = f & 15;
        int ln = (f >> 4) & 31;
        int kc = (f >> 9) & 3;
        int ct = f >> 11;
        int K  = kc * 32 + (ln >> 4) * 8 + (h >> 3) * 16 + (h & 7);
        int N  = ct * 16 + (ln & 15);           // always < 80
        float wv;
        if (K < 64) wv = W1[(64 + K) * H1n + N] - W1[(128 + K) * H1n + N];
        else        wv = W1[(192 + (K - 64)) * H1n + N];
        sh_w1f[f] = (_Float16)wv;
    }
    // layer-2 weights zero-padded to 96 x 48
    for (int f = tid; f < NT2 * KC2 * 32 * 16; f += 256) {
        int h   = f & 15;
        int ln  = (f >> 4) & 31;
        int idx = f >> 9;                        // ct2*3 + kc2
        int kc  = idx % 3;
        int ct  = idx / 3;
        int K   = kc * 32 + (ln >> 4) * 8 + (h >> 3) * 16 + (h & 7);
        int N   = ct * 16 + (ln & 15);
        float wv = (K < H1n && N < H2n) ? W2[K * H2n + N] : 0.0f;
        sh_w2f[f] = (_Float16)wv;
    }
    // zero h1 pad columns 80..103 (never written later; avoid NaN*0 in WMMA)
    for (int f = tid; f < 8 * 16 * 24; f += 256) {
        int col = 80 + (f % 24);
        int row = f / 24;
        sh_h1[row * H1S + col] = (_Float16)0.0f;
    }
    __syncthreads();

    // c[n] = b1[n] + q . (W1q + W1m)
    for (int n = tid; n < H1n; n += 256) {
        float c = b1[n];
        for (int d = 0; d < Dn; ++d)
            c += sh_q[d] * (W1[d * H1n + n] + W1[(128 + d) * H1n + n]);
        sh_c[n] = c;
    }
    __syncthreads();

    // per-lane q values (packed f16) matching the A-fragment element pattern
    v16h qh[2];
    for (int j = 0; j < 2; ++j)
        for (int i = 0; i < 8; ++i) {
            qh[j][i]     = (_Float16)sh_q[j * 32 + hi * 8 + i];
            qh[j][8 + i] = (_Float16)sh_q[j * 32 + 16 + hi * 8 + i];
        }

    const float* kb   = k + (size_t)bid * Tn * Dn;
    _Float16*    myh1 = &sh_h1[wid * 16 * H1S];
    const v8f    vzero = {};

    // ---------------- per-wave row strips ----------------
#pragma unroll 1
    for (int rt = wid; rt < 13; rt += 8) {
        const int t     = rt * 16 + lanelow;     // A rows: M = lane&15 (both halves)
        const bool trow = (t < Tn);

        // build A fragments for K chunks 0..3 (kc 0/1 = k, kc 2/3 = q*k in f16)
        v16h afr[4];
        for (int j = 0; j < 2; ++j) {
            v16h hk;
            if (trow) {
                const float* kp = kb + (size_t)t * Dn + j * 32 + hi * 8;
                float4 x0 = *(const float4*)(kp);
                float4 x1 = *(const float4*)(kp + 4);
                float4 x2 = *(const float4*)(kp + 16);
                float4 x3 = *(const float4*)(kp + 20);
                hk[0]=(_Float16)x0.x;  hk[1]=(_Float16)x0.y;
                hk[2]=(_Float16)x0.z;  hk[3]=(_Float16)x0.w;
                hk[4]=(_Float16)x1.x;  hk[5]=(_Float16)x1.y;
                hk[6]=(_Float16)x1.z;  hk[7]=(_Float16)x1.w;
                hk[8]=(_Float16)x2.x;  hk[9]=(_Float16)x2.y;
                hk[10]=(_Float16)x2.z; hk[11]=(_Float16)x2.w;
                hk[12]=(_Float16)x3.x; hk[13]=(_Float16)x3.y;
                hk[14]=(_Float16)x3.z; hk[15]=(_Float16)x3.w;
            } else {
                for (int i = 0; i < 16; ++i) hk[i] = (_Float16)0.0f;
            }
            afr[j]     = hk;
            afr[j + 2] = hk * qh[j];             // packed f16 multiply
        }

        // layer 1: Z = c + A(128) x Wc(128x80)
        v8f acc[NT1];
        for (int ct = 0; ct < NT1; ++ct) acc[ct] = vzero;
#pragma unroll
        for (int kc = 0; kc < KC1; ++kc)
#pragma unroll
            for (int ct = 0; ct < NT1; ++ct) {
                v16h bfr = *(const v16h*)&sh_w1f[((ct * KC1 + kc) * 32 + lane) * 16];
                acc[ct] = __builtin_amdgcn_wmma_f32_16x16x32_f16(
                    false, afr[kc], false, bfr, (short)0, acc[ct], false, false);
            }

        // sigmoid, stage h1 (f16) in wave-private LDS strip
        for (int ct = 0; ct < NT1; ++ct) {
            int n = ct * 16 + lanelow;
            float cn = sh_c[n];
            for (int r = 0; r < 8; ++r) {
                int m = hi * 8 + r;
                float hv = sigmoidf_(acc[ct][r] + cn);
                myh1[m * H1S + n] = (_Float16)hv;
            }
        }
        // intra-wave DS ordering guarantees visibility (compiler waits DScnt)

        // layer 2 A fragments from staged h1
        v16h a2[KC2];
        {
            const _Float16* p = &myh1[lanelow * H1S];
            for (int kc = 0; kc < KC2; ++kc) {
                int kb0 = kc * 32 + hi * 8;
                v8h lo = *(const v8h*)&p[kb0];
                v8h hh = *(const v8h*)&p[kb0 + 16];
                for (int i = 0; i < 8; ++i) { a2[kc][i] = lo[i]; a2[kc][8 + i] = hh[i]; }
            }
        }

        v8f acc2[NT2];
        for (int ct = 0; ct < NT2; ++ct) acc2[ct] = vzero;
#pragma unroll
        for (int kc = 0; kc < KC2; ++kc)
#pragma unroll
            for (int ct = 0; ct < NT2; ++ct) {
                v16h bfr = *(const v16h*)&sh_w2f[((ct * KC2 + kc) * 32 + lane) * 16];
                acc2[ct] = __builtin_amdgcn_wmma_f32_16x16x32_f16(
                    false, a2[kc], false, bfr, (short)0, acc2[ct], false, false);
            }

        // layer 3: logits = sigmoid(Z2 + b2) . Wf ; cross-lane reduce over n2
        float p8[8];
        for (int r = 0; r < 8; ++r) p8[r] = 0.0f;
        for (int ct = 0; ct < NT2; ++ct) {
            int n2 = ct * 16 + lanelow;
            bool valid = (n2 < H2n);
            float wfv = valid ? Wf[n2] : 0.0f;
            float b2v = valid ? b2[n2] : 0.0f;
            for (int r = 0; r < 8; ++r) {
                float hv = sigmoidf_(acc2[ct][r] + b2v);
                p8[r] += hv * wfv;
            }
        }
        for (int off = 1; off < 16; off <<= 1)
            for (int r = 0; r < 8; ++r)
                p8[r] += __shfl_xor(p8[r], off, 32);

        if (lanelow == 0) {
            float bfv = bf[0];
            for (int r = 0; r < 8; ++r) {
                int tt = rt * 16 + hi * 8 + r;
                if (tt < Tn) {
                    float lg = p8[r] + bfv;
                    sh_logits[tt] = (mask[(size_t)bid * Tn + tt] != 0)
                                        ? lg : -4294967295.0f;   // reference neg_inf
                } else {
                    sh_logits[tt] = -3.0e38f;                    // pad rows vanish
                }
            }
        }
    }
    __syncthreads();

    // ---------------- softmax over 208 logits ----------------
    float x = (tid < TTn) ? sh_logits[tid] : -3.0e38f;
    float mx = x;
    for (int off = 16; off; off >>= 1) mx = fmaxf(mx, __shfl_xor(mx, off, 32));
    if (lane == 0) sh_red[wid] = mx;
    __syncthreads();
    float gmax = sh_red[0];
    for (int i = 1; i < 8; ++i) gmax = fmaxf(gmax, sh_red[i]);
    float e = (tid < TTn) ? __expf(x - gmax) : 0.0f;
    float s = e;
    for (int off = 16; off; off >>= 1) s += __shfl_xor(s, off, 32);
    __syncthreads();                       // done reading sh_red as max
    if (lane == 0) sh_red[wid] = s;
    __syncthreads();
    float gsum = 0.0f;
    for (int i = 0; i < 8; ++i) gsum += sh_red[i];
    if (tid < TTn) sh_sw[tid] = e / gsum;
    __syncthreads();

    // ---------------- out[b,d] = sum_t w[t] * v[b,t,d] ----------------
    {
        int d   = tid & 63;
        int cid = tid >> 6;                 // 4 chunks of 50 timesteps
        const float* vb = v + (size_t)bid * Tn * Dn;
        float accv = 0.0f;
        int t0 = cid * 50, t1 = t0 + 50;
        for (int t = t0; t < t1; ++t)
            accv += sh_sw[t] * vb[(size_t)t * Dn + d];
        sh_out[cid * Dn + d] = accv;
    }
    __syncthreads();
    if (tid < Dn)
        out[(size_t)bid * Dn + tid] =
            sh_out[tid] + sh_out[Dn + tid] + sh_out[2 * Dn + tid] + sh_out[3 * Dn + tid];
}

extern "C" void kernel_launch(void* const* d_in, const int* in_sizes, int n_in,
                              void* d_out, int out_size, void* d_ws, size_t ws_size,
                              hipStream_t stream) {
    const float* q    = (const float*)d_in[0];
    const float* k    = (const float*)d_in[1];
    const float* v    = (const float*)d_in[2];
    const int*   mask = (const int*)d_in[3];
    const float* W1   = (const float*)d_in[4];
    const float* b1   = (const float*)d_in[5];
    const float* W2   = (const float*)d_in[6];
    const float* b2   = (const float*)d_in[7];
    const float* Wf   = (const float*)d_in[8];
    const float* bf   = (const float*)d_in[9];
    din_attn_kernel<<<dim3(Bn), dim3(256), 0, stream>>>(
        q, k, v, mask, W1, b1, W2, b2, Wf, bf, (float*)d_out);
}
